// CTCLoss_70961449664599
// MI455X (gfx1250) — compile-verified, matrix operand order checked by reading, not run on hardware
//
#include <hip/hip_runtime.h>

typedef __attribute__((ext_vector_type(16))) _Float16     v16h;
typedef __attribute__((ext_vector_type(8)))  float        v8f;
typedef __attribute__((ext_vector_type(4)))  unsigned int u32x4;
typedef __attribute__((ext_vector_type(8)))  int          i32x8;
typedef __attribute__((ext_vector_type(4)))  int          i32x4;

#define CTC_NEG (-1e30f)

// ---------------------------------------------------------------------------
// Kernel 1: per-frame logsumexp over the vocab dim, using WMMA-with-ones as a
// cross-column reduction engine. Each wave handles 16 rows (frames) of V=512.
// ---------------------------------------------------------------------------
__global__ __launch_bounds__(256) void lse_wmma_kernel(
    const float* __restrict__ in, float* __restrict__ lse, int nrows, int V)
{
    const int lane    = threadIdx.x & 31;
    const int wave    = threadIdx.x >> 5;
    const int rowBase = (blockIdx.x * 8 + wave) * 16;   // 8 waves / block
    if (rowBase + 16 > nrows) return;                   // wave-uniform guard

    const int   m     = lane & 15;   // A-matrix row owned by this lane
    const int   half  = lane >> 4;   // K-half selector (ISA A layout)
    const float SHIFT = 4.0f;

    const float* rp = in + (size_t)(rowBase + m) * V;

    v16h ones;
#pragma unroll
    for (int i = 0; i < 16; ++i) ones[i] = (_Float16)1.0f;

    v8f acc = {0.f, 0.f, 0.f, 0.f, 0.f, 0.f, 0.f, 0.f};

    for (int k0 = 0; k0 < V; k0 += 32) {
        if (k0 + 64 < V) __builtin_prefetch(rp + k0 + 64, 0, 0);
        v16h a;
#pragma unroll
        for (int j = 0; j < 8; ++j) {
            // 16-bit A 16x32 layout: lanes 0-15 hold K 0..7,16..23; lanes 16-31 hold K 8..15,24..31
            const int kk = k0 + half * 8 + ((j < 4) ? (2 * j) : (16 + 2 * (j - 4)));
            const float2 x = *reinterpret_cast<const float2*>(rp + kk);
            a[2 * j]     = (_Float16)__expf(x.x - SHIFT);
            a[2 * j + 1] = (_Float16)__expf(x.y - SHIFT);
        }
        // D = A * ones + C  ->  per-row running sum of exp()
        acc = __builtin_amdgcn_wmma_f32_16x16x32_f16(
            /*neg_a=*/false, a, /*neg_b=*/false, ones,
            /*c_mod=*/(short)0, acc, /*reuse_a=*/false, /*reuse_b=*/false);
    }

    // D layout: lane 0 (n=0) holds rows 0..7 in acc[0..7]; lane 16 holds rows 8..15.
    if (m == 0) {
        const int rb = rowBase + half * 8;
#pragma unroll
        for (int j = 0; j < 8; ++j)
            lse[rb + j] = __logf(acc[j]) + SHIFT;
    }
}

// ---------------------------------------------------------------------------
// Tensor Data Mover: DMA one row of `nelem` f32 (global -> LDS).
// D# built per CDNA5 ISA:
//   group0: count=1 | lds_addr | global_addr[56:0] | type=2
//   group1: data_size=4B, tensor_dim0=nelem, tensor_dim1=1,
//           tile_dim0=nelem, tile_dim1=1, tensor_dim0_stride=nelem
//   groups 2/3 (+ trailing group): zero (tile is 2-D)
// ---------------------------------------------------------------------------
__device__ __forceinline__ void tdm_load_row(const float* gptr,
                                             unsigned lds_addr, int nelem)
{
    const unsigned long long ga = (unsigned long long)(size_t)gptr;
    u32x4 g0;
    g0[0] = 1u;                                               // count = 1
    g0[1] = lds_addr;                                         // LDS byte address
    g0[2] = (unsigned)ga;                                     // global_addr[31:0]
    g0[3] = (unsigned)((ga >> 32) & 0x1FFFFFFull) | (2u << 30); // [56:32] | type=2

    i32x8 g1;
    g1[0] = (int)(2u << 16);                    // wg_mask=0, data_size=2 (4B)
    g1[1] = (int)((unsigned)nelem << 16);       // tensor_dim0[15:0] @ bits 63:48
    g1[2] = (int)(1u << 16);                    // tensor_dim1=1     @ bits 95:80
    g1[3] = (int)((unsigned)nelem << 16);       // tile_dim0         @ bits 127:112
    g1[4] = 1;                                  // tile_dim1=1       @ bits 143:128
    g1[5] = nelem;                              // tensor_dim0_stride[31:0]
    g1[6] = 0;
    g1[7] = 0;

    i32x4 z4 = {0, 0, 0, 0};
    i32x8 z8 = {0, 0, 0, 0, 0, 0, 0, 0};
    // clang-23 / therock-10.0 6-arg form: (g0, g1, g2, g3, gx, cpol)
    __builtin_amdgcn_tensor_load_to_lds(g0, g1, z4, z4, z8, 0);
}

// ---------------------------------------------------------------------------
// Kernel 2: CTC alpha recursion. One block per batch element, one thread per
// extended-label position l in [0, L=2S+1). Alpha double-buffered in LDS ->
// ONE barrier per time step. Probability rows staged by the TDM into a
// double-buffered LDS tile, DMA for row t+1 overlapping the step-t update.
// ---------------------------------------------------------------------------
__global__ __launch_bounds__(544) void ctc_alpha_kernel(
    const float* __restrict__ in, const int* __restrict__ targets,
    const int* __restrict__ in_len, const int* __restrict__ tgt_len,
    const float* __restrict__ lse, float* __restrict__ loss_out,
    int T, int B, int V, int S)
{
    constexpr int LMAX = 513;   // 2*256 + 1
    constexpr int VMAX = 512;
    __shared__ float aBuf[2][LMAX];
    __shared__ float row[2][VMAX];

    const int  b      = blockIdx.x;
    const int  tid    = threadIdx.x;
    const int  L      = 2 * S + 1;
    const bool active = tid < L;

    // LDS byte addresses of the two row buffers (flat LDS addr, low 32 bits)
    const unsigned ldsRow0 = (unsigned)(size_t)(void*)&row[0][0];
    const unsigned ldsRow1 = (unsigned)(size_t)(void*)&row[1][0];

    // extended label for this position; skip-transition mask
    int  extl   = 0;       // even l -> blank (0)
    bool allow2 = false;
    if (active && (tid & 1)) {
        extl = targets[b * S + (tid >> 1)];
        if (tid >= 3) allow2 = (extl != targets[b * S + ((tid - 2) >> 1)]);
    }
    const int Tb = in_len[b];

    // wave 0: DMA row t=0 into row[0] and wait for it
    if (tid < 32) {
        tdm_load_row(in + (size_t)b * V, ldsRow0, V);
        __builtin_amdgcn_s_wait_tensorcnt(0);
    }
    __syncthreads();            // row 0 visible to all waves

    // alpha init at t=0
    float a = CTC_NEG;
    if (active) {
        const float lp0 = row[0][extl] - lse[b];
        a = (tid < 2) ? lp0 : CTC_NEG;
        aBuf[0][tid] = a;
    }

    // wave 0: kick off DMA for row t=1 (overlaps nothing yet, lands in row[1])
    if (tid < 32 && T > 1)
        tdm_load_row(in + (size_t)(B + b) * V, ldsRow1, V);

    int curIdx = 0;
    for (int t = 1; t < T; ++t) {
        const int pb = t & 1;
        if (tid < 32)
            __builtin_amdgcn_s_wait_tensorcnt(0);   // row t DMA complete
        __syncthreads();   // row t visible; prev-iter alpha/row reads done (WAR)

        // kick off DMA for row t+1 into the buffer whose readers just finished
        if (tid < 32 && t + 1 < T)
            tdm_load_row(in + (size_t)((t + 1) * B + b) * V,
                         (pb == 0) ? ldsRow1 : ldsRow0, V);

        const int nxtIdx = curIdx ^ 1;
        if (active) {
            const float lp  = row[pb][extl] - lse[t * B + b];
            const float a0  = a;
            const float am1 = (tid >= 1) ? aBuf[curIdx][tid - 1] : CTC_NEG;
            const float am2 = allow2    ? aBuf[curIdx][tid - 2] : CTC_NEG;
            const float mx  = fmaxf(a0, fmaxf(am1, am2));
            const float s   = __expf(a0 - mx) + __expf(am1 - mx) + __expf(am2 - mx);
            const float na  = mx + __logf(s) + lp;
            if (t < Tb) a = na;          // freeze padded frames (t >= input_length)
            aBuf[nxtIdx][tid] = a;
        }
        curIdx = nxtIdx;
    }

    __syncthreads();
    if (tid == 0) {
        const int   tl = tgt_len[b];
        const float al = aBuf[curIdx][2 * tl];
        const float ap = aBuf[curIdx][2 * tl - 1];
        const float mx = fmaxf(al, ap);
        float loss = -(mx + __logf(__expf(al - mx) + __expf(ap - mx)));
        if (isinf(loss) || loss > 0.5e30f) loss = 0.0f;   // zero_infinity
        loss_out[b] = loss / (float)tl;                   // per-batch normalized
    }
}

// ---------------------------------------------------------------------------
// Kernel 3: scalar mean over batch.
// ---------------------------------------------------------------------------
__global__ void ctc_reduce_kernel(const float* __restrict__ loss_ws,
                                  float* __restrict__ out, int B)
{
    if (threadIdx.x == 0 && blockIdx.x == 0) {
        float s = 0.0f;
        for (int i = 0; i < B; ++i) s += loss_ws[i];
        out[0] = s / (float)B;
    }
}

extern "C" void kernel_launch(void* const* d_in, const int* in_sizes, int n_in,
                              void* d_out, int out_size, void* d_ws, size_t ws_size,
                              hipStream_t stream)
{
    const float* in      = (const float*)d_in[0];
    const int*   targets = (const int*)d_in[1];
    const int*   in_len  = (const int*)d_in[2];
    const int*   tgt_len = (const int*)d_in[3];
    float*       out     = (float*)d_out;

    const int B = in_sizes[2];            // 64
    const int S = in_sizes[1] / B;        // 256
    const int V = 512;
    const int T = in_sizes[0] / (B * V);  // 2048

    float* lse     = (float*)d_ws;                 // T*B floats
    float* loss_ws = lse + (size_t)T * B;          // B floats

    const int nrows = T * B;                       // 131072, multiple of 128
    lse_wmma_kernel<<<(nrows + 127) / 128, 256, 0, stream>>>(in, lse, nrows, V);
    ctc_alpha_kernel<<<B, 544, 0, stream>>>(in, targets, in_len, tgt_len,
                                            lse, loss_ws, T, B, V, S);
    ctc_reduce_kernel<<<1, 64, 0, stream>>>(loss_ws, out, B);
}
